// RelPartialLearnableMultiHeadAttn_40364102648069
// MI455X (gfx1250) — matrix operand order repaired and checked by reading, not compiled, not run on hardware
//
#include <hip/hip_runtime.h>

// ---------------------------------------------------------------------------
// Types for CDNA5 WMMA (gfx1250, wave32)
// ---------------------------------------------------------------------------
typedef __bf16 bf16_t;
typedef __attribute__((ext_vector_type(16))) __bf16 v16bf;
typedef __attribute__((ext_vector_type(8)))  __bf16 v8bf;
typedef __attribute__((ext_vector_type(8)))  float  v8f;

#define QLEN   1024
#define BSZ    4
#define DMODEL 1024
#define NHEAD  16
#define DHEAD  64
#define MROWS  (QLEN * BSZ)        // 4096
#define NEGV   (-1.0e30f)

__device__ __forceinline__ v16bf cat8(v8bf lo, v8bf hi) {
    union { v16bf v; v8bf h[2]; } u;
    u.h[0] = lo; u.h[1] = hi;
    return u.v;
}
__device__ __forceinline__ v8f zero8() {
    v8f z = {0.f, 0.f, 0.f, 0.f, 0.f, 0.f, 0.f, 0.f};
    return z;
}
__device__ __forceinline__ bf16_t f2bf(float x) { return (bf16_t)x; }

// LDS byte offset of a shared-memory pointer (generic ptr low 32 bits == LDS
// offset within the wave's allocation; hardware adds LDS_BASE for async ops).
__device__ __forceinline__ unsigned lds_off(const void* p) {
    return (unsigned)(unsigned long long)p;
}

// gfx1250 async tensor path: global -> LDS direct, tracked by ASYNCcnt.
__device__ __forceinline__ void async_ld_b128(unsigned lds, const void* g) {
    asm volatile("global_load_async_to_lds_b128 %0, %1, off"
                 :: "v"(lds), "v"(g) : "memory");
}
__device__ __forceinline__ void wait_async4() {
    asm volatile("s_wait_asynccnt 4" ::: "memory");
}
__device__ __forceinline__ void wait_async0() {
    asm volatile("s_wait_asynccnt 0" ::: "memory");
}

// ---------------------------------------------------------------------------
// Elementwise helpers
// ---------------------------------------------------------------------------
__global__ void conv_f32_bf16(const float* __restrict__ S, bf16_t* __restrict__ D, int n) {
    int g = blockIdx.x * blockDim.x + threadIdx.x;
    if (g < n) D[g] = f2bf(S[g]);
}

// src [R][C] f32 -> dst [C][R] bf16 (weight transpose so WMMA B-frags are contiguous)
__global__ void transpose_conv(const float* __restrict__ S, bf16_t* __restrict__ D, int R, int C) {
    int g = blockIdx.x * blockDim.x + threadIdx.x;
    if (g >= R * C) return;
    int c = g % C, r = g / C;
    D[(size_t)c * R + r] = f2bf(S[g]);
}

// ---------------------------------------------------------------------------
// bf16 WMMA GEMM:  C[M][N] = A[M][K] * BT[N][K]^T   (fp32 accumulate)
// 128x128 tile, 256 threads = 8 waves (4x2), 2x4 16x16 frags per wave.
// Double-buffered LDS fill via GLOBAL_LOAD_ASYNC_TO_LDS_B128 (ASYNCcnt),
// overlapping the next tile's global->LDS DMA with the current tile's WMMAs.
// ---------------------------------------------------------------------------
__global__ __launch_bounds__(256)
void gemm_bf16_nt(const bf16_t* __restrict__ A, const bf16_t* __restrict__ BT,
                  float* __restrict__ C, int M, int N, int K) {
    __shared__ __align__(16) bf16_t sA[2][128 * 32];
    __shared__ __align__(16) bf16_t sB[2][128 * 32];

    const int tid  = threadIdx.x;
    const int lane = tid & 31, wid = tid >> 5;
    const int wm = wid & 3, wn = wid >> 2;         // 4 row-stripes x 2 col-stripes
    const int l16 = lane & 15, h = lane >> 4;
    const int m0 = blockIdx.x * 128, n0 = blockIdx.y * 128;

    // per-thread fill coordinates (4 async b128 per thread per tile: 2 A + 2 B)
    const int fr0 = tid >> 2,         fk0 = (tid & 3) * 8;          // chunk 0
    const int fr1 = (tid + 256) >> 2, fk1 = ((tid + 256) & 3) * 8;  // chunk 1
    const unsigned aOff0 = lds_off(&sA[0][0]), aOff1 = lds_off(&sA[1][0]);
    const unsigned bOff0 = lds_off(&sB[0][0]), bOff1 = lds_off(&sB[1][0]);

    v8f acc[2][4];
#pragma unroll
    for (int mt = 0; mt < 2; ++mt)
#pragma unroll
        for (int nt = 0; nt < 4; ++nt) acc[mt][nt] = zero8();

    auto issue_fill = [&](unsigned aOff, unsigned bOff, int kb) {
        async_ld_b128(aOff + (unsigned)(fr0 * 32 + fk0) * 2,
                      &A [(size_t)(m0 + fr0) * K + kb + fk0]);
        async_ld_b128(bOff + (unsigned)(fr0 * 32 + fk0) * 2,
                      &BT[(size_t)(n0 + fr0) * K + kb + fk0]);
        async_ld_b128(aOff + (unsigned)(fr1 * 32 + fk1) * 2,
                      &A [(size_t)(m0 + fr1) * K + kb + fk1]);
        async_ld_b128(bOff + (unsigned)(fr1 * 32 + fk1) * 2,
                      &BT[(size_t)(n0 + fr1) * K + kb + fk1]);
    };

    issue_fill(aOff0, bOff0, 0);                   // prologue: tile 0 in flight

    int p = 0;
    for (int kb = 0; kb < K; kb += 32, p ^= 1) {
        const bool more = (kb + 32) < K;
        if (more) issue_fill(p ? aOff0 : aOff1, p ? bOff0 : bOff1, kb + 32);
        if (more) wait_async4(); else wait_async0();   // current buffer landed
        __syncthreads();                                // visible to all waves

        v16bf af[2], bfr[4];
#pragma unroll
        for (int mt = 0; mt < 2; ++mt) {
            int m = wm * 32 + mt * 16 + l16;
            af[mt] = cat8(*(const v8bf*)(&sA[p][m * 32 + h * 8]),
                          *(const v8bf*)(&sA[p][m * 32 + 16 + h * 8]));
        }
#pragma unroll
        for (int nt = 0; nt < 4; ++nt) {
            int n = wn * 64 + nt * 16 + l16;
            bfr[nt] = cat8(*(const v8bf*)(&sB[p][n * 32 + h * 8]),
                           *(const v8bf*)(&sB[p][n * 32 + 16 + h * 8]));
        }
#pragma unroll
        for (int mt = 0; mt < 2; ++mt)
#pragma unroll
            for (int nt = 0; nt < 4; ++nt)
                acc[mt][nt] = __builtin_amdgcn_wmma_f32_16x16x32_bf16(
                    false, af[mt], false, bfr[nt], (short)0, acc[mt][nt], false, false);
        __syncthreads();                                // done reading buf p
    }

#pragma unroll
    for (int mt = 0; mt < 2; ++mt)
#pragma unroll
        for (int nt = 0; nt < 4; ++nt) {
            int col = n0 + wn * 64 + nt * 16 + l16;
#pragma unroll
            for (int v = 0; v < 8; ++v) {
                int row = m0 + wm * 32 + mt * 16 + v + 8 * h;
                C[(size_t)row * N + col] = acc[mt][nt][v];
            }
        }
}

// ---------------------------------------------------------------------------
// Prep: heads[4096][3072] f32 -> bf16 QAC/QBD (biases folded), K [b][n][j][d],
//       VT [b][n][d][j] (transposed so PV B-frags are contiguous)
// ---------------------------------------------------------------------------
__global__ void prep_qkv(const float* __restrict__ heads,
                         const float* __restrict__ rwb, const float* __restrict__ rrb,
                         bf16_t* __restrict__ QAC, bf16_t* __restrict__ QBD,
                         bf16_t* __restrict__ Kb,  bf16_t* __restrict__ VT) {
    int g = blockIdx.x * blockDim.x + threadIdx.x;
    if (g >= BSZ * NHEAD * QLEN * DHEAD) return;
    int d = g & 63;
    int i = (g >> 6) & 1023;
    int n = (g >> 16) & 15;
    int b = g >> 20;
    size_t hrow = (size_t)(i * BSZ + b) * (3 * DMODEL);
    float q = heads[hrow + n * 64 + d];
    float k = heads[hrow + DMODEL + n * 64 + d];
    float v = heads[hrow + 2 * DMODEL + n * 64 + d];
    size_t head = (size_t)(b * NHEAD + n);
    size_t idx  = (head * QLEN + i) * DHEAD + d;
    QAC[idx] = f2bf(q + rwb[n * 64 + d]);
    QBD[idx] = f2bf(q + rrb[n * 64 + d]);
    Kb[idx]  = f2bf(k);
    VT[(head * DHEAD + d) * QLEN + i] = f2bf(v);
}

// rk[p][n*64+d] f32 -> RK[n][p][d] bf16
__global__ void prep_rk(const float* __restrict__ rk, bf16_t* __restrict__ RK) {
    int g = blockIdx.x * blockDim.x + threadIdx.x;
    if (g >= NHEAD * QLEN * DHEAD) return;
    int d = g & 63;
    int p = (g >> 6) & 1023;
    int n = g >> 16;
    RK[((size_t)n * QLEN + p) * DHEAD + d] = f2bf(rk[(size_t)p * DMODEL + n * 64 + d]);
}

// ---------------------------------------------------------------------------
// Flash attention with Transformer-XL relative shift.
// grid(16 itiles, 16 heads, 4 batch), 128 threads = 4 independent waves.
// Each wave owns 16 query rows; per 64-key tile: 8 WMMA AC + 10 WMMA banded BD
// + online softmax in wave-private LDS + 8 WMMA PV.
// rel_shift (causal region): BD[i][j] = qbd_i . rk[1023 + j - i]
// ---------------------------------------------------------------------------
__global__ __launch_bounds__(128)
void attn_flash(const bf16_t* __restrict__ QAC, const bf16_t* __restrict__ QBD,
                const bf16_t* __restrict__ Kb,  const bf16_t* __restrict__ VT,
                const bf16_t* __restrict__ RK,  float* __restrict__ vec) {
    __shared__ __align__(16) float sS [4][16 * 64];
    __shared__ __align__(16) float sBD[4][16 * 80];
    __shared__ float sM[4][16], sL[4][16], sA[4][16];

    const int tid  = threadIdx.x;
    const int lane = tid & 31, w = tid >> 5;
    const int l16 = lane & 15, h = lane >> 4;
    const int n = blockIdx.y, b = blockIdx.z;
    const int i0w = blockIdx.x * 64 + w * 16;      // wave's first query row

    const size_t head = (size_t)(b * NHEAD + n);
    const bf16_t* Qac = QAC + (head * QLEN + i0w) * DHEAD;
    const bf16_t* Qbd = QBD + (head * QLEN + i0w) * DHEAD;
    const bf16_t* Kh  = Kb + head * QLEN * DHEAD;
    const bf16_t* Vt  = VT + head * DHEAD * QLEN;
    const bf16_t* Rn  = RK + (size_t)n * QLEN * DHEAD;

    // Q fragments (rows = l16, two K halves of d)
    v16bf qac[2], qbd[2];
#pragma unroll
    for (int kh = 0; kh < 2; ++kh) {
        const bf16_t* ba = Qac + (size_t)l16 * DHEAD + kh * 32;
        const bf16_t* bb = Qbd + (size_t)l16 * DHEAD + kh * 32;
        qac[kh] = cat8(*(const v8bf*)(ba + h * 8), *(const v8bf*)(ba + 16 + h * 8));
        qbd[kh] = cat8(*(const v8bf*)(bb + h * 8), *(const v8bf*)(bb + 16 + h * 8));
    }

    v8f o[4];
#pragma unroll
    for (int dt = 0; dt < 4; ++dt) o[dt] = zero8();
    if (lane < 16) { sM[w][l16] = -3.0e38f; sL[w][l16] = 0.f; }

    float* S  = sS[w];
    float* BD = sBD[w];
    const float scale = 0.125f;                    // 1/sqrt(64)
    const int jend = i0w + 15;

    for (int j0 = 0; j0 <= jend; j0 += 64) {
        // ---- AC = (q + r_w_bias) K^T  -> LDS S[16][64]
#pragma unroll
        for (int nt = 0; nt < 4; ++nt) {
            v8f acc = zero8();
            int j = j0 + nt * 16 + l16;
#pragma unroll
            for (int kh = 0; kh < 2; ++kh) {
                const bf16_t* kp = Kh + (size_t)j * DHEAD + kh * 32;
                v16bf bf = cat8(*(const v8bf*)(kp + h * 8), *(const v8bf*)(kp + 16 + h * 8));
                acc = __builtin_amdgcn_wmma_f32_16x16x32_bf16(
                    false, qac[kh], false, bf, (short)0, acc, false, false);
            }
#pragma unroll
            for (int v = 0; v < 8; ++v) S[(v + 8 * h) * 64 + nt * 16 + l16] = acc[v];
        }
        // ---- BD band: raw[i][cb] = qbd_i . rk[pbase + cb], width 80
        int pbase = 1023 + j0 - i0w - 15;
#pragma unroll
        for (int pt = 0; pt < 5; ++pt) {
            v8f acc = zero8();
            int p = pbase + pt * 16 + l16;
            p = (p > 1023) ? 1023 : p;             // masked cols anyway
#pragma unroll
            for (int kh = 0; kh < 2; ++kh) {
                const bf16_t* rp = Rn + (size_t)p * DHEAD + kh * 32;
                v16bf bf = cat8(*(const v8bf*)(rp + h * 8), *(const v8bf*)(rp + 16 + h * 8));
                acc = __builtin_amdgcn_wmma_f32_16x16x32_bf16(
                    false, qbd[kh], false, bf, (short)0, acc, false, false);
            }
#pragma unroll
            for (int v = 0; v < 8; ++v) BD[(v + 8 * h) * 80 + pt * 16 + l16] = acc[v];
        }

        // ---- online softmax: lane handles row l16, cols [h*32, h*32+32)
        int   rr   = l16;
        int   irow = i0w + rr;
        float mOld = sM[w][rr];
        float lmax = -3.0e38f;
        for (int cc = 0; cc < 32; ++cc) {
            int c = h * 32 + cc;
            int j = j0 + c;
            float s = (j <= irow)
                    ? (S[rr * 64 + c] + BD[rr * 80 + (c - rr + 15)]) * scale
                    : NEGV;
            S[rr * 64 + c] = s;
            lmax = fmaxf(lmax, s);
        }
        lmax = fmaxf(lmax, __shfl_xor(lmax, 16));
        float mNew  = fmaxf(mOld, lmax);
        float alpha = __expf(mOld - mNew);
        float lsum = 0.f;
        for (int cc = 0; cc < 32; ++cc) {
            int c = h * 32 + cc;
            float pe = __expf(S[rr * 64 + c] - mNew);
            S[rr * 64 + c] = pe;
            lsum += pe;
        }
        lsum += __shfl_xor(lsum, 16);
        if (lane < 16) {
            sM[w][rr] = mNew;
            sL[w][rr] = sL[w][rr] * alpha + lsum;
            sA[w][rr] = alpha;
        }

        // ---- rescale accumulators
#pragma unroll
        for (int v = 0; v < 8; ++v) {
            float a = sA[w][v + 8 * h];
#pragma unroll
            for (int dt = 0; dt < 4; ++dt) o[dt][v] *= a;
        }

        // ---- P fragments (f32 LDS -> bf16 A-frag layout)
        v16bf pf[2];
#pragma unroll
        for (int kh = 0; kh < 2; ++kh) {
            const float* pr = &S[l16 * 64 + kh * 32];
            v16bf t;
#pragma unroll
            for (int e = 0; e < 8; ++e) t[e]     = f2bf(pr[h * 8 + e]);
#pragma unroll
            for (int e = 0; e < 8; ++e) t[e + 8] = f2bf(pr[16 + h * 8 + e]);
            pf[kh] = t;
        }
        // ---- O += P @ V
#pragma unroll
        for (int dt = 0; dt < 4; ++dt) {
            int d = dt * 16 + l16;
#pragma unroll
            for (int kh = 0; kh < 2; ++kh) {
                const bf16_t* vp = Vt + (size_t)d * QLEN + j0 + kh * 32;
                v16bf vf = cat8(*(const v8bf*)(vp + h * 8), *(const v8bf*)(vp + 16 + h * 8));
                o[dt] = __builtin_amdgcn_wmma_f32_16x16x32_bf16(
                    false, pf[kh], false, vf, (short)0, o[dt], false, false);
            }
        }
    }

    // ---- normalize, write vec[(i*4+b)][n*64+d]
    float linv[8];
#pragma unroll
    for (int v = 0; v < 8; ++v) linv[v] = 1.0f / sL[w][v + 8 * h];
#pragma unroll
    for (int dt = 0; dt < 4; ++dt) {
#pragma unroll
        for (int v = 0; v < 8; ++v) {
            int irow = i0w + v + 8 * h;
            int col  = n * 64 + dt * 16 + l16;
            vec[(size_t)(irow * BSZ + b) * DMODEL + col] = o[dt][v] * linv[v];
        }
    }
}

// ---------------------------------------------------------------------------
// Residual + LayerNorm: out = LN(w + attn_out) * gamma + beta, one row/block
// ---------------------------------------------------------------------------
__global__ __launch_bounds__(256)
void ln_residual(const float* __restrict__ w, const float* __restrict__ ao,
                 const float* __restrict__ gamma, const float* __restrict__ beta,
                 float* __restrict__ out) {
    const int row = blockIdx.x;
    const float* xw = w  + (size_t)row * DMODEL;
    const float* xa = ao + (size_t)row * DMODEL;
    float* op = out + (size_t)row * DMODEL;
    __shared__ float red[8];
    const int tid = threadIdx.x, lane = tid & 31, wv = tid >> 5;

    float x[4];
    float s = 0.f;
#pragma unroll
    for (int e = 0; e < 4; ++e) { x[e] = xw[tid + e * 256] + xa[tid + e * 256]; s += x[e]; }
#pragma unroll
    for (int off = 16; off > 0; off >>= 1) s += __shfl_xor(s, off);
    if (lane == 0) red[wv] = s;
    __syncthreads();
    float tot = 0.f;
#pragma unroll
    for (int k = 0; k < 8; ++k) tot += red[k];
    const float mean = tot * (1.0f / DMODEL);
    __syncthreads();

    float vs = 0.f;
#pragma unroll
    for (int e = 0; e < 4; ++e) { float d = x[e] - mean; vs += d * d; }
#pragma unroll
    for (int off = 16; off > 0; off >>= 1) vs += __shfl_xor(vs, off);
    if (lane == 0) red[wv] = vs;
    __syncthreads();
    float tv = 0.f;
#pragma unroll
    for (int k = 0; k < 8; ++k) tv += red[k];
    const float rstd = rsqrtf(tv * (1.0f / DMODEL) + 1e-5f);
#pragma unroll
    for (int e = 0; e < 4; ++e) {
        int c = tid + e * 256;
        op[c] = (x[e] - mean) * rstd * gamma[c] + beta[c];
    }
}

// ---------------------------------------------------------------------------
// Host orchestration
// ---------------------------------------------------------------------------
extern "C" void kernel_launch(void* const* d_in, const int* in_sizes, int n_in,
                              void* d_out, int out_size, void* d_ws, size_t ws_size,
                              hipStream_t stream) {
    (void)in_sizes; (void)n_in; (void)out_size; (void)ws_size;
    const float* w    = (const float*)d_in[0];
    const float* r    = (const float*)d_in[1];
    const float* rwb  = (const float*)d_in[2];
    const float* rrb  = (const float*)d_in[3];
    // d_in[4]: attn_mask — causal tril, derived analytically (j <= i)
    const float* Wqkv = (const float*)d_in[5];
    const float* Wr   = (const float*)d_in[6];
    const float* Wo   = (const float*)d_in[7];
    const float* gm   = (const float*)d_in[8];
    const float* bt   = (const float*)d_in[9];
    float* out = (float*)d_out;

    char* ws = (char*)d_ws;
    size_t off = 0;
    auto take = [&](size_t bytes) -> void* {
        off = (off + 255) & ~(size_t)255;
        void* p = ws + off;
        off += bytes;
        return p;
    };

    bf16_t* WqkvT = (bf16_t*)take((size_t)3 * DMODEL * DMODEL * 2);
    bf16_t* WrT   = (bf16_t*)take((size_t)DMODEL * DMODEL * 2);
    bf16_t* WoT   = (bf16_t*)take((size_t)DMODEL * DMODEL * 2);
    bf16_t* w_bf  = (bf16_t*)take((size_t)MROWS * DMODEL * 2);
    bf16_t* r_bf  = (bf16_t*)take((size_t)QLEN * DMODEL * 2);
    float*  heads = (float*) take((size_t)MROWS * 3 * DMODEL * 4);
    float*  rk    = (float*) take((size_t)QLEN * DMODEL * 4);
    bf16_t* QACb  = (bf16_t*)take((size_t)BSZ * NHEAD * QLEN * DHEAD * 2);
    bf16_t* QBDb  = (bf16_t*)take((size_t)BSZ * NHEAD * QLEN * DHEAD * 2);
    bf16_t* Kbf   = (bf16_t*)take((size_t)BSZ * NHEAD * QLEN * DHEAD * 2);
    bf16_t* VTb   = (bf16_t*)take((size_t)BSZ * NHEAD * QLEN * DHEAD * 2);
    bf16_t* RKb   = (bf16_t*)take((size_t)NHEAD * QLEN * DHEAD * 2);
    float*  vec   = (float*) take((size_t)MROWS * DMODEL * 4);
    bf16_t* vecbf = (bf16_t*)take((size_t)MROWS * DMODEL * 2);
    float*  aout  = (float*) take((size_t)MROWS * DMODEL * 4);

    const int TPB = 256;
    // weight transposes / input conversions
    transpose_conv<<<(DMODEL * 3 * DMODEL + TPB - 1) / TPB, TPB, 0, stream>>>(Wqkv, WqkvT, DMODEL, 3 * DMODEL);
    transpose_conv<<<(DMODEL * DMODEL + TPB - 1) / TPB, TPB, 0, stream>>>(Wr, WrT, DMODEL, DMODEL);
    transpose_conv<<<(DMODEL * DMODEL + TPB - 1) / TPB, TPB, 0, stream>>>(Wo, WoT, DMODEL, DMODEL);
    conv_f32_bf16<<<(MROWS * DMODEL + TPB - 1) / TPB, TPB, 0, stream>>>(w, w_bf, MROWS * DMODEL);
    conv_f32_bf16<<<(QLEN * DMODEL + TPB - 1) / TPB, TPB, 0, stream>>>(r, r_bf, QLEN * DMODEL);

    // projections
    gemm_bf16_nt<<<dim3(MROWS / 128, (3 * DMODEL) / 128), 256, 0, stream>>>(
        w_bf, WqkvT, heads, MROWS, 3 * DMODEL, DMODEL);
    gemm_bf16_nt<<<dim3(QLEN / 128, DMODEL / 128), 256, 0, stream>>>(
        r_bf, WrT, rk, QLEN, DMODEL, DMODEL);

    // head-major repack + bias folding
    prep_qkv<<<(BSZ * NHEAD * QLEN * DHEAD + TPB - 1) / TPB, TPB, 0, stream>>>(
        heads, rwb, rrb, QACb, QBDb, Kbf, VTb);
    prep_rk<<<(NHEAD * QLEN * DHEAD + TPB - 1) / TPB, TPB, 0, stream>>>(rk, RKb);

    // flash attention with relative shift
    attn_flash<<<dim3(QLEN / 64, NHEAD, BSZ), 128, 0, stream>>>(
        QACb, QBDb, Kbf, VTb, RKb, vec);

    // output projection + residual layernorm
    conv_f32_bf16<<<(MROWS * DMODEL + TPB - 1) / TPB, TPB, 0, stream>>>(vec, vecbf, MROWS * DMODEL);
    gemm_bf16_nt<<<dim3(MROWS / 128, DMODEL / 128), 256, 0, stream>>>(
        vecbf, WoT, aout, MROWS, DMODEL, DMODEL);
    ln_residual<<<MROWS, 256, 0, stream>>>(w, aout, gm, bt, out);
}
